// MoE_3367254360090
// MI455X (gfx1250) — matrix operand order repaired and checked by reading, not compiled, not run on hardware
//
#include <hip/hip_runtime.h>
#include <hip/hip_bf16.h>
#include <math.h>

#define D_MODEL 1024
#define D_FF    2816
#define N_EXP   8
#define N_TOK   4096
#define TILE_T  64
#define FC      256
#define N_CHUNK (D_FF / FC)   // 11
#define NWAVES  16
#define BLOCK   (NWAVES * 32) // 512
#define XROW    (D_MODEL + 8) // padded LDS row stride (bank-conflict-free)
#define HROW    (FC + 8)      // padded LDS row stride for h tile

typedef __bf16 bf16;
typedef __attribute__((ext_vector_type(16))) __bf16 v16bf;
typedef __attribute__((ext_vector_type(8)))  __bf16 v8bf;
typedef __attribute__((ext_vector_type(8)))  float  v8f;

// ---------------------------------------------------------------- helpers ---

__device__ __forceinline__ v8f v8f_zero() {
  v8f z;
#pragma unroll
  for (int i = 0; i < 8; ++i) z[i] = 0.0f;
  return z;
}

__device__ __forceinline__ v8f wmma_bf16(v16bf a, v16bf b, v8f c) {
  // D = A(16x32 bf16) * B(32x16 bf16) + C(16x16 f32)
  return __builtin_amdgcn_wmma_f32_16x16x32_bf16(false, a, false, b,
                                                 (short)0, c, false, false);
}

// A-matrix fragment (16x32 bf16): lane L holds row M=L%16.
// lanes 0-15: K = kb+0..15 ; lanes 16-31: K = kb+16..31 (contiguous)
__device__ __forceinline__ v16bf load_a_row(const bf16* row, int kb, int hi) {
  v8bf lo = *(const v8bf*)(row + kb + hi * 16);
  v8bf hh = *(const v8bf*)(row + kb + hi * 16 + 8);
  v16bf a;
#pragma unroll
  for (int i = 0; i < 8; ++i) { a[i] = lo[i]; a[i + 8] = hh[i]; }
  return a;
}

// B-matrix fragment (32x16 bf16): lane L holds column N=L%16 = one weight row.
// lanes 0-15: K = kb+0..15 ; lanes 16-31: K = kb+16..31 (contiguous -> float4s)
__device__ __forceinline__ v16bf load_b_row_f32(const float* row, int kb, int hi) {
  const float4* p = (const float4*)(row + kb + hi * 16);
  float4 f0 = p[0], f1 = p[1], f2 = p[2], f3 = p[3];
  v16bf b;
  b[0]  = (bf16)f0.x; b[1]  = (bf16)f0.y; b[2]  = (bf16)f0.z; b[3]  = (bf16)f0.w;
  b[4]  = (bf16)f1.x; b[5]  = (bf16)f1.y; b[6]  = (bf16)f1.z; b[7]  = (bf16)f1.w;
  b[8]  = (bf16)f2.x; b[9]  = (bf16)f2.y; b[10] = (bf16)f2.z; b[11] = (bf16)f2.w;
  b[12] = (bf16)f3.x; b[13] = (bf16)f3.y; b[14] = (bf16)f3.z; b[15] = (bf16)f3.w;
  return b;
}

__device__ __forceinline__ float gelu_exact(float x) {
  return 0.5f * x * (1.0f + erff(x * 0.70710678118654752f));
}

// ----------------------------------------------------------------- kernels --

__global__ void k_init(int* counts) {
  if (threadIdx.x < N_EXP) counts[threadIdx.x] = 0;
}

// One wave32 per token: RMSNorm + gate softmax top-2 + bf16 xn + expert lists.
__global__ __launch_bounds__(256)
void k_norm_gate(const float* __restrict__ x, const float* __restrict__ gw,
                 bf16* __restrict__ xn, int* __restrict__ counts,
                 int* __restrict__ tidx, float* __restrict__ tw) {
  int wave = threadIdx.x >> 5, lane = threadIdx.x & 31;
  int t = blockIdx.x * 8 + wave;
  const float* xr = x + (size_t)t * D_MODEL;
  int kb = lane * 32;

  float v[32];
  float ss = 0.0f;
#pragma unroll
  for (int i = 0; i < 8; ++i) {
    float4 f = *(const float4*)(xr + kb + i * 4);
    v[i * 4 + 0] = f.x; v[i * 4 + 1] = f.y; v[i * 4 + 2] = f.z; v[i * 4 + 3] = f.w;
    ss += f.x * f.x + f.y * f.y + f.z * f.z + f.w * f.w;
  }
  float dot[N_EXP];
#pragma unroll
  for (int e = 0; e < N_EXP; ++e) {
    const float* g = gw + (size_t)e * D_MODEL + kb;
    float d = 0.0f;
#pragma unroll
    for (int i = 0; i < 8; ++i) {
      float4 f = *(const float4*)(g + i * 4);
      d += f.x * v[i * 4 + 0] + f.y * v[i * 4 + 1] +
           f.z * v[i * 4 + 2] + f.w * v[i * 4 + 3];
    }
    dot[e] = d;
  }
  // wave32 reductions
#pragma unroll
  for (int off = 16; off > 0; off >>= 1) {
    ss += __shfl_xor(ss, off, 32);
#pragma unroll
    for (int e = 0; e < N_EXP; ++e) dot[e] += __shfl_xor(dot[e], off, 32);
  }
  float rn = rsqrtf(ss * (1.0f / D_MODEL) + 1e-6f);

  // write normalized activations as bf16
  bf16* xo = xn + (size_t)t * D_MODEL + kb;
#pragma unroll
  for (int i = 0; i < 4; ++i) {
    v8bf o;
#pragma unroll
    for (int j = 0; j < 8; ++j) o[j] = (bf16)(v[i * 8 + j] * rn);
    *(v8bf*)(xo + i * 8) = o;
  }

  if (lane == 0) {
    float l[N_EXP], mx = -1e30f;
#pragma unroll
    for (int e = 0; e < N_EXP; ++e) { l[e] = dot[e] * rn; mx = fmaxf(mx, l[e]); }
    float p[N_EXP];
#pragma unroll
    for (int e = 0; e < N_EXP; ++e) p[e] = __expf(l[e] - mx);
    int i1 = 0;
#pragma unroll
    for (int e = 1; e < N_EXP; ++e) if (p[e] > p[i1]) i1 = e;
    int i2 = (i1 == 0) ? 1 : 0;
#pragma unroll
    for (int e = 0; e < N_EXP; ++e) if (e != i1 && p[e] > p[i2]) i2 = e;
    float inv = 1.0f / (p[i1] + p[i2]);   // softmax denom cancels in renorm
    int pa = atomicAdd(&counts[i1], 1);
    tidx[i1 * N_TOK + pa] = t; tw[i1 * N_TOK + pa] = p[i1] * inv;
    int pb = atomicAdd(&counts[i2], 1);
    tidx[i2 * N_TOK + pb] = t; tw[i2 * N_TOK + pb] = p[i2] * inv;
  }
}

// Fused FFN over a 64-token tile: out += cw * ( W2^T @ (gelu(W1 x) * (W3 x)) ).
// The gathered activation tile (chunk-invariant A matrix) is staged ONCE into
// LDS with GLOBAL_LOAD_ASYNC_TO_LDS_B128 (ASYNCcnt), leaving the global-load
// path entirely to the 1.24 GB weight stream. Row strides padded so 16-row
// column reads rotate across LDS banks (no 16-way conflicts).
// MODE 0: routed expert (gather tokens, gate-weighted atomic add)
// MODE 1: shared expert over all tokens (writes acc + identity, initializes out)
template <int MODE>
__global__ __launch_bounds__(BLOCK, 1)
void k_ffn(const bf16* __restrict__ xn,
           const float* __restrict__ W1, const float* __restrict__ W3,
           const float* __restrict__ W2,
           const int* __restrict__ counts, const int* __restrict__ tidx,
           const float* __restrict__ tw,
           const float* __restrict__ x_id, float* __restrict__ out) {
  __shared__ bf16  sh_x[TILE_T][XROW];  // 129 KB staged activations (async)
  __shared__ bf16  sh_h[TILE_T][HROW];  // 33 KB staged intermediate
  __shared__ int   s_tok[TILE_T];
  __shared__ float s_tw[TILE_T];

  int e = 0, n = N_TOK, base = blockIdx.x * TILE_T;
  if (MODE == 0) {
    e = blockIdx.y;
    n = counts[e];
    if (base >= n) return;
  }
  const float* W1e = W1 + (size_t)e * D_FF * D_MODEL;
  const float* W3e = W3 + (size_t)e * D_FF * D_MODEL;
  const float* W2e = W2 + (size_t)e * D_MODEL * D_FF;

  if (threadIdx.x < TILE_T) {
    int i = base + threadIdx.x;
    if (MODE == 0) {
      bool ok = i < n;
      s_tok[threadIdx.x] = ok ? tidx[e * N_TOK + i] : 0;
      s_tw[threadIdx.x]  = ok ? tw[e * N_TOK + i] : 0.0f;
    } else {
      s_tok[threadIdx.x] = i;
      s_tw[threadIdx.x]  = 1.0f;
    }
  }
  __syncthreads();

  // ---- async-gather the 64x1024 bf16 activation tile into LDS ----
  // 8192 x 16B chunks, 512 threads -> 16 async B128 copies per thread.
  {
    unsigned lds0 = (unsigned)(uintptr_t)&sh_x[0][0];
#pragma unroll
    for (int i = 0; i < 16; ++i) {
      int cid = i * BLOCK + threadIdx.x;     // 0..8191
      int row = cid >> 7;                    // 128 chunks per 2KB row
      int c16 = cid & 127;
      unsigned long long g = (unsigned long long)(uintptr_t)
          (xn + (size_t)s_tok[row] * D_MODEL + c16 * 8);
      unsigned l = lds0 + (unsigned)(row * (XROW * 2) + c16 * 16);
      asm volatile("global_load_async_to_lds_b128 %0, %1, off"
                   :: "v"(l), "v"(g) : "memory");
    }
    asm volatile("s_wait_asynccnt 0x0" ::: "memory");
  }
  __syncthreads();

  int wave = threadIdx.x >> 5, lane = threadIdx.x & 31;
  int lm = lane & 15, hi = lane >> 4;

  // per-wave output accumulators: 64 tokens x 64 dims = 4 Mblk x 4 Nblk tiles
  v8f acc[4][4];
#pragma unroll
  for (int m = 0; m < 4; ++m)
#pragma unroll
    for (int nb = 0; nb < 4; ++nb) acc[m][nb] = v8f_zero();

  for (int c = 0; c < N_CHUNK; ++c) {
    int f0 = c * FC;
    // ---- phase 1: G1,G3 [64 tok x 16 f] for this wave's f-slice ----
    v8f g1[4], g3[4];
#pragma unroll
    for (int m = 0; m < 4; ++m) { g1[m] = v8f_zero(); g3[m] = v8f_zero(); }
    const float* w1row = W1e + (size_t)(f0 + wave * 16 + lm) * D_MODEL;
    const float* w3row = W3e + (size_t)(f0 + wave * 16 + lm) * D_MODEL;
    for (int k = 0; k < D_MODEL; k += 32) {
      // load all A fragments first: ds_loads overlap, no WMMA->VALU WAR bubbles
      v16bf a[4];
#pragma unroll
      for (int m = 0; m < 4; ++m)
        a[m] = load_a_row(&sh_x[16 * m + lm][0], k, hi);  // ds_load_b128 x2
      v16bf b1 = load_b_row_f32(w1row, k, hi);
      v16bf b3 = load_b_row_f32(w3row, k, hi);
#pragma unroll
      for (int m = 0; m < 4; ++m) g1[m] = wmma_bf16(a[m], b1, g1[m]);
#pragma unroll
      for (int m = 0; m < 4; ++m) g3[m] = wmma_bf16(a[m], b3, g3[m]);
    }
    // prefetch next chunk's W1/W3 rows for this wave (overlaps phase 2/3);
    // locality 3 -> WGP scope: pull into all cache levels
    if (c + 1 < N_CHUNK) {
      const float* nw1 = w1row + (size_t)FC * D_MODEL;
      const float* nw3 = w3row + (size_t)FC * D_MODEL;
#pragma unroll
      for (int q = 0; q < 8; ++q) {
        __builtin_prefetch(nw1 + q * 128, 0, 3);
        __builtin_prefetch(nw3 + q * 128, 0, 3);
      }
    }
    // ---- phase 2: h = gelu(G1) * G3 -> LDS (bf16) ----
    __syncthreads();   // previous chunk's phase-3 readers done
#pragma unroll
    for (int m = 0; m < 4; ++m)
#pragma unroll
      for (int r = 0; r < 8; ++r) {
        int t = 16 * m + r + hi * 8;
        sh_h[t][wave * 16 + lm] = (bf16)(gelu_exact(g1[m][r]) * g3[m][r]);
      }
    __syncthreads();
    // prefetch next chunk's W2 segments for this wave's dims
    if (c + 1 < N_CHUNK) {
#pragma unroll
      for (int nb = 0; nb < 4; ++nb) {
        const float* nw2 = W2e + (size_t)(wave * 64 + nb * 16 + lm) * D_FF
                          + f0 + FC;
#pragma unroll
        for (int q = 0; q < 8; ++q) __builtin_prefetch(nw2 + q * 32, 0, 3);
      }
    }
    // ---- phase 3: acc += h[64 x 256] @ W2[:, chunk]^T (wave's 64 dims) ----
    for (int k = 0; k < FC; k += 32) {
      v16bf a[4];
#pragma unroll
      for (int m = 0; m < 4; ++m)
        a[m] = load_a_row(&sh_h[16 * m + lm][0], k, hi);
#pragma unroll
      for (int nb = 0; nb < 4; ++nb) {
        int dim = wave * 64 + nb * 16 + lm;
        const float* w2row = W2e + (size_t)dim * D_FF + f0;
        v16bf b = load_b_row_f32(w2row, k, hi);
#pragma unroll
        for (int m = 0; m < 4; ++m) acc[m][nb] = wmma_bf16(a[m], b, acc[m][nb]);
      }
    }
  }

  // ---- epilogue ----
#pragma unroll
  for (int m = 0; m < 4; ++m) {
#pragma unroll
    for (int r = 0; r < 8; ++r) {
      int tl = 16 * m + r + hi * 8;
      int tok = s_tok[tl];
      float cw = s_tw[tl];
#pragma unroll
      for (int nb = 0; nb < 4; ++nb) {
        int dim = wave * 64 + nb * 16 + lm;
        size_t o = (size_t)tok * D_MODEL + dim;
        float val = acc[m][nb][r];
        if (MODE == 0) {
          atomicAdd(&out[o], val * cw);
        } else {
          out[o] = val + x_id[o];   // shared expert + residual (initializes out)
        }
      }
    }
  }
}

// ------------------------------------------------------------------ launch --

extern "C" void kernel_launch(void* const* d_in, const int* in_sizes, int n_in,
                              void* d_out, int out_size, void* d_ws, size_t ws_size,
                              hipStream_t stream) {
  const float* x   = (const float*)d_in[0];
  const float* gw  = (const float*)d_in[1];
  const float* w1  = (const float*)d_in[2];
  const float* w2  = (const float*)d_in[3];
  const float* w3  = (const float*)d_in[4];
  const float* sw1 = (const float*)d_in[5];
  const float* sw2 = (const float*)d_in[6];
  const float* sw3 = (const float*)d_in[7];
  float* out = (float*)d_out;
  (void)in_sizes; (void)n_in; (void)out_size; (void)ws_size;

  char* ws = (char*)d_ws;
  size_t off = 0;
  bf16* xn    = (bf16*)(ws + off);  off += (size_t)N_TOK * D_MODEL * sizeof(bf16);
  int* counts = (int*)(ws + off);   off += 64;
  int* tidx   = (int*)(ws + off);   off += (size_t)N_EXP * N_TOK * sizeof(int);
  float* tw   = (float*)(ws + off); off += (size_t)N_EXP * N_TOK * sizeof(float);

  k_init<<<1, 64, 0, stream>>>(counts);
  k_norm_gate<<<N_TOK / 8, 256, 0, stream>>>(x, gw, xn, counts, tidx, tw);
  // shared expert + residual first (plain stores initialize d_out) ...
  k_ffn<1><<<N_TOK / TILE_T, BLOCK, 0, stream>>>(xn, sw1, sw3, sw2,
                                                 counts, tidx, tw, x, out);
  // ... then routed experts accumulate atomically
  k_ffn<0><<<dim3(N_TOK / TILE_T, N_EXP), BLOCK, 0, stream>>>(xn, w1, w3, w2,
                                                              counts, tidx, tw, x, out);
}
